// MultimodalNetwork_77498389889515
// MI455X (gfx1250) — matrix operand order repaired and checked by reading, not compiled, not run on hardware
//
#include <hip/hip_runtime.h>
#include <hip/hip_bf16.h>

// ---------------------------------------------------------------------------
// Types for CDNA5 WMMA (wave32): D(16x16 f32) = A(16x32 bf16) x B(32x16 bf16) + C
// ---------------------------------------------------------------------------
typedef __attribute__((ext_vector_type(16))) __bf16 v16bf;
typedef __attribute__((ext_vector_type(8)))  __bf16 v8bf;
typedef __attribute__((ext_vector_type(8)))  float  v8f;

#define T_SEQ 256
#define HD    32          // LSTM hidden size == WMMA K
#define BN_EPS 1e-5f

__device__ __forceinline__ v8f wmma_bf16(v16bf a, v16bf b, v8f c) {
  // (neg_a, A, neg_b, B, c_mod, C, reuse_a, reuse_b)
  return __builtin_amdgcn_wmma_f32_16x16x32_bf16(false, a, false, b, (short)0, c,
                                                 false, false);
}

// A matrix 16x32 bf16, LDS stored row-major [m][k], m=0..15, k=0..31.
// Layout used here: lane(l): M = l%16, element e -> K = e + 16*(l/16) (16 contiguous k, matches B).
__device__ __forceinline__ v16bf load_A_frag(const __bf16* lds) {
  int lane = threadIdx.x & 31;
  int m    = lane & 15;
  int base = (lane >> 4) << 4;   // 0 or 16
  return *(const v16bf*)(lds + m * HD + base);
}

// B matrix 32x16 bf16 for gate tile tn. W stored in LDS row-major [gate][k] (128x32).
// Math: gates[m][gc] = sum_k h[m][k] * W[gc][k]  => B[K][N] = W[16*tn + N][K].
// Layout used here: lane(l): N = l%16, element e -> K = e + 16*(l/16) (16 contiguous k).
__device__ __forceinline__ v16bf load_B_frag(const __bf16* ldsW, int tn) {
  int lane  = threadIdx.x & 31;
  int n     = lane & 15;
  int khalf = (lane >> 4) << 4;  // 0 or 16
  return *(const v16bf*)(ldsW + (tn * 16 + n) * HD + khalf);
}

__device__ __forceinline__ float sigmoidf_(float x) { return 1.f / (1.f + __expf(-x)); }

// ---------------------------------------------------------------------------
// Forward layer-0 LSTM: one wave per tile of 16 sequences, scans t=0..T-1,
// streams hs_f[n][t][0..31] (f32) to workspace.  Gates via WMMA:
//   G = h @ Whh^T + xpad @ WihPad^T + b   (xpad: d_in zero-padded to K=32)
// All 16 weight B-fragments hoisted into registers (loop-invariant).
// ---------------------------------------------------------------------------
__global__ __launch_bounds__(32, 1) void lstm_fwd_kernel(
    const float* __restrict__ x, long row_stride, int dimraw, int d_in,
    const float* __restrict__ Wih, const float* __restrict__ Whh,
    const float* __restrict__ bias, float* __restrict__ hsf)
{
  __shared__ __align__(64) __bf16 sWhh[128 * HD];
  __shared__ __align__(64) __bf16 sWih[128 * HD];
  __shared__ __align__(64) __bf16 sH[16 * HD];
  __shared__ __align__(64) __bf16 sX[16 * HD];
  __shared__ int sLen[16];

  const int lane = threadIdx.x;
  const int n0   = blockIdx.x * 16;

  for (int i = lane; i < 128 * HD; i += 32) {
    int gc = i >> 5, k = i & 31;
    sWhh[i] = (__bf16)Whh[gc * HD + k];
    sWih[i] = (__bf16)((k < d_in) ? Wih[gc * d_in + k] : 0.f);
  }
  for (int i = lane; i < 16 * HD; i += 32) sH[i] = (__bf16)0.f;
  if (lane < 16)
    sLen[lane] = (int)x[(long)(n0 + lane) * row_stride + (row_stride - 1)];
  __syncthreads();

  // hoist weight fragments into registers
  v16bf Bh[8], Bx[8];
#pragma unroll
  for (int tn = 0; tn < 8; ++tn) {
    Bh[tn] = load_B_frag(sWhh, tn);
    Bx[tn] = load_B_frag(sWih, tn);
  }

  const int col = lane & 15, hi = lane >> 4;
  float breg[8];
#pragma unroll
  for (int tn = 0; tn < 8; ++tn) breg[tn] = bias[tn * 16 + col];
  int lenreg[8];
#pragma unroll
  for (int r = 0; r < 8; ++r) lenreg[r] = sLen[r + 8 * hi];

  float cst[2][8], hst[2][8];
#pragma unroll
  for (int u = 0; u < 2; ++u)
#pragma unroll
    for (int r = 0; r < 8; ++r) { cst[u][r] = 0.f; hst[u][r] = 0.f; }

  for (int t = 0; t < T_SEQ; ++t) {
    { // stage x_t tile, zero-padded to K=32 (lane l covers seq m=l/2, k half l%2)
      int m = lane >> 1, k0 = (lane & 1) << 4;
      const float* xrow = x + (long)(n0 + m) * row_stride + (long)t * dimraw;
#pragma unroll
      for (int j = 0; j < 16; ++j) {
        int k = k0 + j;
        sX[m * HD + k] = (__bf16)((k < d_in) ? xrow[k] : 0.f);
      }
    }
    __syncthreads();

    v16bf aH = load_A_frag(sH);
    v16bf aX = load_A_frag(sX);
    v8f acc[8];
#pragma unroll
    for (int tn = 0; tn < 8; ++tn) {
      v8f c;
#pragma unroll
      for (int r = 0; r < 8; ++r) c[r] = breg[tn];
      c = wmma_bf16(aH, Bh[tn], c);
      c = wmma_bf16(aX, Bx[tn], c);
      acc[tn] = c;
    }

    // gate order (torch): i=0..31, f=32..63, g=64..95, o=96..127
#pragma unroll
    for (int u = 0; u < 2; ++u)
#pragma unroll
      for (int r = 0; r < 8; ++r) {
        float ig = sigmoidf_(acc[u][r]);
        float fg = sigmoidf_(acc[u + 2][r]);
        float gg = tanhf(acc[u + 4][r]);
        float og = sigmoidf_(acc[u + 6][r]);
        float cn = fg * cst[u][r] + ig * gg;
        float hn = og * tanhf(cn);
        if (t < lenreg[r]) { cst[u][r] = cn; hst[u][r] = hn; }
      }

    __syncthreads();
#pragma unroll
    for (int u = 0; u < 2; ++u)
#pragma unroll
      for (int r = 0; r < 8; ++r) {
        int m  = r + 8 * hi;
        int hc = u * 16 + col;
        sH[m * HD + hc] = (__bf16)hst[u][r];
        hsf[((long)(n0 + m) * T_SEQ + t) * HD + hc] = hst[u][r];
      }
    __syncthreads();
  }
}

// ---------------------------------------------------------------------------
// Fused reverse scan: layer-0 backward + layer-1 backward in one loop.
// At step s (pos = len-1-s while valid):
//   L0b: G0 = h0 @ Whh0^T + xpad[pos] @ Wih0Pad^T + b0  -> new h0 (== hs_b[pos])
//   L1b: G1 = h1 @ Whh1^T + hs_f[pos] @ Wih1[:, :32]^T + h0 @ Wih1[:, 32:]^T + b1
// Final h1 (frozen at each sequence's last valid step) * mask -> out.
// All 40 weight B-fragments hoisted into registers (wave32 VGPR file is large).
// ---------------------------------------------------------------------------
__global__ __launch_bounds__(32, 1) void lstm_bwd_kernel(
    const float* __restrict__ x, long row_stride, int dimraw, int d_in,
    const float* __restrict__ Wih0, const float* __restrict__ Whh0, const float* __restrict__ b0,
    const float* __restrict__ Wih1, const float* __restrict__ Whh1, const float* __restrict__ b1,
    const float* __restrict__ hsf, const float* __restrict__ mask,
    int seq_per_b, int mask_cols, float* __restrict__ outp)
{
  __shared__ __align__(64) __bf16 sWhh0[128 * HD];
  __shared__ __align__(64) __bf16 sWih0[128 * HD];
  __shared__ __align__(64) __bf16 sWhh1[128 * HD];
  __shared__ __align__(64) __bf16 sWih1a[128 * HD];
  __shared__ __align__(64) __bf16 sWih1b[128 * HD];
  __shared__ __align__(64) __bf16 sH0[16 * HD];
  __shared__ __align__(64) __bf16 sH1[16 * HD];
  __shared__ __align__(64) __bf16 sX[16 * HD];
  __shared__ __align__(64) __bf16 sHf[16 * HD];
  __shared__ int sLen[16];

  const int lane = threadIdx.x;
  const int n0   = blockIdx.x * 16;

  for (int i = lane; i < 128 * HD; i += 32) {
    int gc = i >> 5, k = i & 31;
    sWhh0[i]  = (__bf16)Whh0[gc * HD + k];
    sWih0[i]  = (__bf16)((k < d_in) ? Wih0[gc * d_in + k] : 0.f);
    sWhh1[i]  = (__bf16)Whh1[gc * HD + k];
    sWih1a[i] = (__bf16)Wih1[gc * 64 + k];
    sWih1b[i] = (__bf16)Wih1[gc * 64 + 32 + k];
  }
  for (int i = lane; i < 16 * HD; i += 32) { sH0[i] = (__bf16)0.f; sH1[i] = (__bf16)0.f; }
  if (lane < 16)
    sLen[lane] = (int)x[(long)(n0 + lane) * row_stride + (row_stride - 1)];
  __syncthreads();

  // hoist all weight fragments into registers (loop-invariant)
  v16bf Bh0[8], Bx0[8], Bh1[8], Bf1[8], Bb1[8];
#pragma unroll
  for (int tn = 0; tn < 8; ++tn) {
    Bh0[tn] = load_B_frag(sWhh0,  tn);
    Bx0[tn] = load_B_frag(sWih0,  tn);
    Bh1[tn] = load_B_frag(sWhh1,  tn);
    Bf1[tn] = load_B_frag(sWih1a, tn);
    Bb1[tn] = load_B_frag(sWih1b, tn);
  }

  const int col = lane & 15, hi = lane >> 4;
  float breg0[8], breg1[8];
#pragma unroll
  for (int tn = 0; tn < 8; ++tn) { breg0[tn] = b0[tn * 16 + col]; breg1[tn] = b1[tn * 16 + col]; }
  int lenreg[8];
#pragma unroll
  for (int r = 0; r < 8; ++r) lenreg[r] = sLen[r + 8 * hi];

  float c0[2][8], h0[2][8], c1[2][8], h1[2][8];
#pragma unroll
  for (int u = 0; u < 2; ++u)
#pragma unroll
    for (int r = 0; r < 8; ++r) { c0[u][r] = 0.f; h0[u][r] = 0.f; c1[u][r] = 0.f; h1[u][r] = 0.f; }

  for (int s = 0; s < T_SEQ; ++s) {
    { // stage reversed x and hs_f gather (hs_f row is 64B-aligned contiguous)
      int m = lane >> 1, k0 = (lane & 1) << 4;
      int lm = sLen[m];
      int p  = (s < lm) ? (lm - 1 - s) : s;
      const float* xrow = x + (long)(n0 + m) * row_stride + (long)p * dimraw;
#pragma unroll
      for (int j = 0; j < 16; ++j) {
        int k = k0 + j;
        sX[m * HD + k] = (__bf16)((k < d_in) ? xrow[k] : 0.f);
      }
      const float4* frow = (const float4*)(hsf + ((long)(n0 + m) * T_SEQ + p) * HD + k0);
#pragma unroll
      for (int j4 = 0; j4 < 4; ++j4) {
        float4 fv = frow[j4];
        int k = k0 + j4 * 4;
        sHf[m * HD + k + 0] = (__bf16)fv.x;
        sHf[m * HD + k + 1] = (__bf16)fv.y;
        sHf[m * HD + k + 2] = (__bf16)fv.z;
        sHf[m * HD + k + 3] = (__bf16)fv.w;
      }
    }
    __syncthreads();

    // ---- layer-0 backward cell ----
    v16bf aH0 = load_A_frag(sH0);
    v16bf aX  = load_A_frag(sX);
    v8f acc[8];
#pragma unroll
    for (int tn = 0; tn < 8; ++tn) {
      v8f c;
#pragma unroll
      for (int r = 0; r < 8; ++r) c[r] = breg0[tn];
      c = wmma_bf16(aH0, Bh0[tn], c);
      c = wmma_bf16(aX,  Bx0[tn], c);
      acc[tn] = c;
    }
#pragma unroll
    for (int u = 0; u < 2; ++u)
#pragma unroll
      for (int r = 0; r < 8; ++r) {
        float ig = sigmoidf_(acc[u][r]);
        float fg = sigmoidf_(acc[u + 2][r]);
        float gg = tanhf(acc[u + 4][r]);
        float og = sigmoidf_(acc[u + 6][r]);
        float cn = fg * c0[u][r] + ig * gg;
        float hn = og * tanhf(cn);
        if (s < lenreg[r]) { c0[u][r] = cn; h0[u][r] = hn; }
      }
    __syncthreads();
#pragma unroll
    for (int u = 0; u < 2; ++u)
#pragma unroll
      for (int r = 0; r < 8; ++r)
        sH0[(r + 8 * hi) * HD + u * 16 + col] = (__bf16)h0[u][r];
    __syncthreads();

    // ---- layer-1 backward cell (input = [hs_f[pos], h0_new]) ----
    v16bf aH1  = load_A_frag(sH1);
    v16bf aHf  = load_A_frag(sHf);
    v16bf aH0n = load_A_frag(sH0);
#pragma unroll
    for (int tn = 0; tn < 8; ++tn) {
      v8f c;
#pragma unroll
      for (int r = 0; r < 8; ++r) c[r] = breg1[tn];
      c = wmma_bf16(aH1,  Bh1[tn], c);
      c = wmma_bf16(aHf,  Bf1[tn], c);
      c = wmma_bf16(aH0n, Bb1[tn], c);
      acc[tn] = c;
    }
#pragma unroll
    for (int u = 0; u < 2; ++u)
#pragma unroll
      for (int r = 0; r < 8; ++r) {
        float ig = sigmoidf_(acc[u][r]);
        float fg = sigmoidf_(acc[u + 2][r]);
        float gg = tanhf(acc[u + 4][r]);
        float og = sigmoidf_(acc[u + 6][r]);
        float cn = fg * c1[u][r] + ig * gg;
        float hn = og * tanhf(cn);
        if (s < lenreg[r]) { c1[u][r] = cn; h1[u][r] = hn; }
      }
    __syncthreads();
#pragma unroll
    for (int u = 0; u < 2; ++u)
#pragma unroll
      for (int r = 0; r < 8; ++r)
        sH1[(r + 8 * hi) * HD + u * 16 + col] = (__bf16)h1[u][r];
    __syncthreads();
  }

  // masked final hidden -> out[n][0..31]
#pragma unroll
  for (int u = 0; u < 2; ++u)
#pragma unroll
    for (int r = 0; r < 8; ++r) {
      int n   = n0 + r + 8 * hi;
      int b   = n / seq_per_b;
      int idx = n % seq_per_b;
      float mv = mask[(long)b * mask_cols + idx];
      outp[(long)n * HD + u * 16 + col] = h1[u][r] * mv;
    }
}

// ---------------------------------------------------------------------------
// 1x1-conv / BN / ReLU head + constant-1 padding.  One thread per output cell
// of [B, 1, 6, 16]; cells with r==5 or c==15 are the appended ones.
// ---------------------------------------------------------------------------
__global__ __launch_bounds__(96) void conv_head_kernel(
    const float* __restrict__ outv, const float* __restrict__ outf,
    const float* __restrict__ w1, const float* __restrict__ b1,
    const float* __restrict__ w2, const float* __restrict__ b2,
    const float* __restrict__ w3, const float* __restrict__ b3,
    const float* __restrict__ w4, const float* __restrict__ b4,
    const float* __restrict__ w5, const float* __restrict__ b5,
    const float* __restrict__ g1, const float* __restrict__ be1,
    const float* __restrict__ m1, const float* __restrict__ v1,
    const float* __restrict__ g2, const float* __restrict__ be2,
    const float* __restrict__ m2, const float* __restrict__ v2,
    const float* __restrict__ g3, const float* __restrict__ be3,
    const float* __restrict__ m3, const float* __restrict__ v3,
    float* __restrict__ out)
{
  int b   = blockIdx.x;
  int tid = threadIdx.x;
  int r = tid >> 4, c = tid & 15;
  float result;
  if (r == 5 || c == 15) {
    result = 1.f;
  } else {
    int ph = r, cam = c;
    float in0[64];
    for (int h = 0; h < HD; ++h) {
      in0[h]      = outf[((long)b * 5  + ph)  * HD + h];   // f_part channels 0..31
      in0[32 + h] = outv[((long)b * 15 + cam) * HD + h];   // v_part channels 32..63
    }
    float t1[128];
    for (int o = 0; o < 128; ++o) {
      float s = b1[o];
      for (int k = 0; k < 64; ++k) s += in0[k] * w1[o * 64 + k];
      s = (s - m1[o]) * (g1[o] / sqrtf(v1[o] + BN_EPS)) + be1[o];
      t1[o] = fmaxf(s, 0.f);
    }
    float t2[64];
    for (int o = 0; o < 64; ++o) {
      float s = b2[o];
      for (int k = 0; k < 128; ++k) s += t1[k] * w2[o * 128 + k];
      s = (s - m2[o]) * (g2[o] / sqrtf(v2[o] + BN_EPS)) + be2[o];
      t2[o] = fmaxf(s, 0.f);
    }
    float t3[32];
    for (int o = 0; o < 32; ++o) {
      float s = b3[o];
      for (int k = 0; k < 64; ++k) s += t2[k] * w3[o * 64 + k];
      s = (s - m3[o]) * (g3[o] / sqrtf(v3[o] + BN_EPS)) + be3[o];
      t3[o] = fmaxf(s, 0.f);
    }
    float t4[16];
    for (int o = 0; o < 16; ++o) {
      float s = b4[o];
      for (int k = 0; k < 32; ++k) s += t3[k] * w4[o * 32 + k];
      t4[o] = fmaxf(s, 0.f);
    }
    float t5 = b5[0];
    for (int k = 0; k < 16; ++k) t5 += t4[k] * w5[k];
    result = fmaxf(t5, 0.f);
  }
  out[(long)b * 96 + tid] = result;
}

// ---------------------------------------------------------------------------
// Host launcher.  Inputs in setup_inputs() dict (insertion) order, nested
// structures flattened in natural order.
// ---------------------------------------------------------------------------
extern "C" void kernel_launch(void* const* d_in, const int* in_sizes, int n_in,
                              void* d_out, int out_size, void* d_ws, size_t ws_size,
                              hipStream_t stream)
{
  (void)in_sizes; (void)n_in; (void)out_size; (void)ws_size;

  const float* x_v        = (const float*)d_in[0];   // [128,15,1281]
  const float* x_f        = (const float*)d_in[1];   // [128,5,2817]
  const float* depth_mask = (const float*)d_in[2];   // [128,16]
  const float* ftm_mask   = (const float*)d_in[3];   // [128,6]
  // lstm_v: l0f(4,5,6) l0b(7,8,9) l1f(10,11,12) l1b(13,14,15)
  const float* v_Wih0f = (const float*)d_in[4];
  const float* v_Whh0f = (const float*)d_in[5];
  const float* v_b0f   = (const float*)d_in[6];
  const float* v_Wih0b = (const float*)d_in[7];
  const float* v_Whh0b = (const float*)d_in[8];
  const float* v_b0b   = (const float*)d_in[9];
  const float* v_Wih1b = (const float*)d_in[13];
  const float* v_Whh1b = (const float*)d_in[14];
  const float* v_b1b   = (const float*)d_in[15];
  // lstm_f: l0f(16,17,18) l0b(19,20,21) l1f(22,23,24) l1b(25,26,27)
  const float* f_Wih0f = (const float*)d_in[16];
  const float* f_Whh0f = (const float*)d_in[17];
  const float* f_b0f   = (const float*)d_in[18];
  const float* f_Wih0b = (const float*)d_in[19];
  const float* f_Whh0b = (const float*)d_in[20];
  const float* f_b0b   = (const float*)d_in[21];
  const float* f_Wih1b = (const float*)d_in[25];
  const float* f_Whh1b = (const float*)d_in[26];
  const float* f_b1b   = (const float*)d_in[27];
  // conv: w/b 28..37, bn1 38..41, bn2 42..45, bn3 46..49
  const float* w1 = (const float*)d_in[28]; const float* b1 = (const float*)d_in[29];
  const float* w2 = (const float*)d_in[30]; const float* b2 = (const float*)d_in[31];
  const float* w3 = (const float*)d_in[32]; const float* b3 = (const float*)d_in[33];
  const float* w4 = (const float*)d_in[34]; const float* b4 = (const float*)d_in[35];
  const float* w5 = (const float*)d_in[36]; const float* b5 = (const float*)d_in[37];
  const float* g1 = (const float*)d_in[38]; const float* be1 = (const float*)d_in[39];
  const float* m1 = (const float*)d_in[40]; const float* v1  = (const float*)d_in[41];
  const float* g2 = (const float*)d_in[42]; const float* be2 = (const float*)d_in[43];
  const float* m2 = (const float*)d_in[44]; const float* v2  = (const float*)d_in[45];
  const float* g3 = (const float*)d_in[46]; const float* be3 = (const float*)d_in[47];
  const float* m3 = (const float*)d_in[48]; const float* v3  = (const float*)d_in[49];

  // Workspace layout (floats): hs_f camera, hs_f phone, out_v, out_f
  float* ws = (float*)d_ws;
  size_t off = 0;
  float* hsf_v = ws + off; off += (size_t)1920 * T_SEQ * HD;  // 15,728,640
  float* hsf_f = ws + off; off += (size_t)640  * T_SEQ * HD;  //  5,242,880
  float* outv  = ws + off; off += (size_t)1920 * HD;
  float* outf  = ws + off; off += (size_t)640  * HD;

  // camera: 1920 seqs / 16 per wave = 120 tiles; phone: 640/16 = 40
  lstm_fwd_kernel<<<120, 32, 0, stream>>>(x_v, 1281L, 5, 3,
                                          v_Wih0f, v_Whh0f, v_b0f, hsf_v);
  lstm_fwd_kernel<<<40, 32, 0, stream>>>(x_f, 2817L, 11, 11,
                                         f_Wih0f, f_Whh0f, f_b0f, hsf_f);
  lstm_bwd_kernel<<<120, 32, 0, stream>>>(x_v, 1281L, 5, 3,
                                          v_Wih0b, v_Whh0b, v_b0b,
                                          v_Wih1b, v_Whh1b, v_b1b,
                                          hsf_v, depth_mask, 15, 16, outv);
  lstm_bwd_kernel<<<40, 32, 0, stream>>>(x_f, 2817L, 11, 11,
                                         f_Wih0b, f_Whh0b, f_b0b,
                                         f_Wih1b, f_Whh1b, f_b1b,
                                         hsf_f, ftm_mask, 5, 6, outf);
  conv_head_kernel<<<128, 96, 0, stream>>>(outv, outf,
                                           w1, b1, w2, b2, w3, b3, w4, b4, w5, b5,
                                           g1, be1, m1, v1,
                                           g2, be2, m2, v2,
                                           g3, be3, m3, v3,
                                           (float*)d_out);
}
